// MyLongformer_86930138071698
// MI455X (gfx1250) — compile-verified, matrix operand order checked by reading, not run on hardware
//
#include <hip/hip_runtime.h>
#include <hip/hip_bf16.h>
#include <stdint.h>

// ---------------- problem constants ----------------
constexpr int B_  = 8;
constexpr int S_  = 4096;
constexpr int D_  = 768;
constexpr int H_  = 12;
constexpr int DH_ = 64;
constexpr int W_  = 128;
constexpr int NB_ = S_ / W_;          // 32
constexpr int L_  = 2;
constexpr int V_  = 512;
constexpr int FF_ = 3072;
constexpr int S2W_ = S_ + 2 * W_;     // 4352
constexpr long MS_ = (long)B_ * S_;   // 32768 rows

typedef __bf16 bf16_t;
typedef __attribute__((ext_vector_type(16))) __bf16 v16bf;
typedef __attribute__((ext_vector_type(8)))  __bf16 v8bf;
typedef __attribute__((ext_vector_type(8)))  float  v8f;
typedef __attribute__((ext_vector_type(4)))  int    v4i;

// CDNA5 async global->LDS copies (ASYNCcnt-tracked), with sync fallback.
#if __has_builtin(__builtin_amdgcn_global_load_async_to_lds_b128) && \
    __has_builtin(__builtin_amdgcn_s_wait_asynccnt)
#define HAVE_ASYNC_LDS 1
#define ASYNC_WAIT(n) __builtin_amdgcn_s_wait_asynccnt(n)
#else
#define HAVE_ASYNC_LDS 0
#define ASYNC_WAIT(n)
#endif

__device__ __forceinline__ void copy16_g2l(const bf16_t* g, bf16_t* l) {
#if HAVE_ASYNC_LDS
  // signature: (v4i addrspace(1)* src, v4i addrspace(3)* dst, imm off, imm cpol)
  __builtin_amdgcn_global_load_async_to_lds_b128(
      (__attribute__((address_space(1))) v4i*)g,
      (__attribute__((address_space(3))) v4i*)l, 0, 0);
#else
  *(v8bf*)l = *(const v8bf*)g;
#endif
}

// ---------------- WMMA helpers ----------------
// A-fragment (16x32 bf16, row-major source, row stride ld elements):
// lane = lo + 16*hi ; row = lo ; elements e<8 -> K = k0 + e + 8*hi,
// e>=8 -> K = k0 + 16 + (e-8) + 8*hi  => two contiguous 16B loads.
// B-fragment mirrored (lane = column), same address math on row-major NxK.
__device__ __forceinline__ v16bf load_frag(const bf16_t* __restrict__ base, int ld) {
  int lane = threadIdx.x & 31;
  int lo = lane & 15, hi = lane >> 4;
  const bf16_t* p = base + (long)lo * ld + 8 * hi;
  v8bf a = *(const v8bf*)(p);
  v8bf b = *(const v8bf*)(p + 16);
  v16bf r;
#pragma unroll
  for (int i = 0; i < 8; ++i) { r[i] = a[i]; r[i + 8] = b[i]; }
  return r;
}

__device__ __forceinline__ v8f wmma_bf16(v16bf a, v16bf b, v8f c) {
  return __builtin_amdgcn_wmma_f32_16x16x32_bf16(false, a, false, b, (short)0, c,
                                                 false, false);
}

// ---------------- small utility kernels ----------------
__global__ void embed_kernel(const int* __restrict__ call,
                             const float* __restrict__ emb,
                             float* __restrict__ xf, bf16_t* __restrict__ xb) {
  long i = (long)blockIdx.x * blockDim.x + threadIdx.x;   // over MS_*D_
  long row = i / D_;
  int  col = (int)(i % D_);
  float v = emb[(long)call[row] * D_ + col];
  xf[i] = v;
  xb[i] = (bf16_t)v;
}

__global__ void cvt_bf16_kernel(const float* __restrict__ src,
                                bf16_t* __restrict__ dst, long n) {
  long i = (long)blockIdx.x * blockDim.x + threadIdx.x;
  if (i < n) dst[i] = (bf16_t)src[i];
}

// keybias: -10000 where att_mask != 0 (global token pos 0, and pads)
// gbias  : -10000 where att_mask < 0 (pads only)
__global__ void bias_kernel(const unsigned char* __restrict__ pad_mask,
                            float* __restrict__ keybias, float* __restrict__ gbias) {
  long i = (long)blockIdx.x * blockDim.x + threadIdx.x;   // over MS_
  int s = (int)(i % S_);
  bool pad = pad_mask[i] != 0;
  float att = pad ? -10000.f : (s == 0 ? 10000.f : 0.f);
  keybias[i] = (att != 0.f) ? -10000.f : 0.f;
  gbias[i]   = (att < 0.f)  ? -10000.f : 0.f;
}

// Build padded K [B,H,S2W,DH] and transposed padded V [B,H,DH,S2W]
__global__ void padkv_kernel(const bf16_t* __restrict__ k_b,
                             const bf16_t* __restrict__ v_b,
                             bf16_t* __restrict__ kpad, bf16_t* __restrict__ vpT) {
  long i = (long)blockIdx.x * blockDim.x + threadIdx.x;   // B*H*S2W*DH
  int d = (int)(i % DH_);
  long t = i / DH_;
  int p = (int)(t % S2W_); t /= S2W_;
  int h = (int)(t % H_);
  int b = (int)(t / H_);
  bf16_t kv = (bf16_t)0.f, vv = (bf16_t)0.f;
  int s = p - W_;
  if (s >= 0 && s < S_) {
    long src = ((long)(b * S_ + s)) * D_ + h * DH_ + d;
    kv = k_b[src];
    vv = v_b[src];
  }
  kpad[i] = kv;
  vpT[(((long)(b * H_ + h) * DH_ + d) * S2W_) + p] = vv;
}

// ---------------- LDS-staged, double-buffered WMMA GEMM ----------------
// C = alpha*(A @ Wt^T + bias), A:[M,K] bf16, Wt:[N,K] bf16, both row-major.
// Block: 256 threads (8 waves) -> 128x64 tile; wave -> 32x32 (2x2 WMMA accs).
// K staged 64 at a time into padded LDS panels (pitch 72 elems: conflict-free
// b128 fragment reads), double-buffered via ASYNCcnt.
constexpr int KP_ = 64;   // K panel depth
constexpr int LP_ = 72;   // padded LDS row pitch (elements)

__global__ void __launch_bounds__(256)
gemm_bf16_kernel(const bf16_t* __restrict__ A, const bf16_t* __restrict__ Wt,
                 const float* __restrict__ bias, float alpha, int relu,
                 float* __restrict__ Cf, bf16_t* __restrict__ Cb,
                 int M, int N, int K) {
  __shared__ bf16_t Ash[2][128 * LP_];   // 2*128*72*2 = 36864 B
  __shared__ bf16_t Bsh[2][64 * LP_];    // 2* 64*72*2 = 18432 B
  int tid  = threadIdx.x;
  int wave = tid >> 5, lane = tid & 31;
  int wm = wave & 3, wn = wave >> 2;     // 4x2 wave grid
  int m0 = blockIdx.x * 128;
  int n0 = blockIdx.y * 64;

  auto stage = [&](int buf, int k) {
    // A panel: 128 rows x 64 elems = 1024 16B chunks, 4 per thread
#pragma unroll
    for (int i = 0; i < 4; ++i) {
      int c = tid + 256 * i;
      int row = c >> 3, col = (c & 7) * 8;
      copy16_g2l(A + (long)(m0 + row) * K + k + col, &Ash[buf][row * LP_ + col]);
    }
    // B panel: 64 rows x 64 elems = 512 chunks, 2 per thread
#pragma unroll
    for (int i = 0; i < 2; ++i) {
      int c = tid + 256 * i;
      int row = c >> 3, col = (c & 7) * 8;
      copy16_g2l(Wt + (long)(n0 + row) * K + k + col, &Bsh[buf][row * LP_ + col]);
    }
  };

  v8f acc00 = {}, acc01 = {}, acc10 = {}, acc11 = {};
  int nst = K / KP_;
  stage(0, 0);
  int buf = 0;
  for (int s = 0; s < nst; ++s) {
    if (s + 1 < nst) {
      stage(buf ^ 1, (s + 1) * KP_);
      ASYNC_WAIT(6);    // 6 async ops of stage s+1 in flight; stage s complete
    } else {
      ASYNC_WAIT(0);
    }
    __syncthreads();
    const bf16_t* ab = &Ash[buf][(wm * 32) * LP_];
    const bf16_t* bb = &Bsh[buf][(wn * 32) * LP_];
#pragma unroll
    for (int kk = 0; kk < KP_; kk += 32) {
      v16bf a0 = load_frag(ab + kk, LP_);
      v16bf a1 = load_frag(ab + 16 * LP_ + kk, LP_);
      v16bf b0 = load_frag(bb + kk, LP_);
      v16bf b1 = load_frag(bb + 16 * LP_ + kk, LP_);
      acc00 = wmma_bf16(a0, b0, acc00);
      acc01 = wmma_bf16(a0, b1, acc01);
      acc10 = wmma_bf16(a1, b0, acc10);
      acc11 = wmma_bf16(a1, b1, acc11);
    }
    __syncthreads();
    buf ^= 1;
  }

  int lo = lane & 15, hi = lane >> 4;
  v8f accs[2][2] = {{acc00, acc01}, {acc10, acc11}};
#pragma unroll
  for (int mi = 0; mi < 2; ++mi) {
#pragma unroll
    for (int ni = 0; ni < 2; ++ni) {
      int col = n0 + wn * 32 + ni * 16 + lo;
      float bv = bias ? bias[col] : 0.f;
#pragma unroll
      for (int r = 0; r < 8; ++r) {
        int row = m0 + wm * 32 + mi * 16 + r + 8 * hi;
        float v = (accs[mi][ni][r] + bv) * alpha;
        if (relu) v = v > 0.f ? v : 0.f;
        long idx = (long)row * N + col;
        if (Cf) Cf[idx] = v;
        if (Cb) Cb[idx] = (bf16_t)v;
      }
    }
  }
}

// ---------------- banded attention ----------------
// One block per (b, h, band-block n, quarter qq): 32 query rows, 384 banded keys
// + 1 global key column; scores via WMMA (K=64), softmax in LDS, out via WMMA (K=384).
__global__ void __launch_bounds__(256)
band_attn_kernel(const bf16_t* __restrict__ q_b,   // [B,S,H*DH] scaled
                 const bf16_t* __restrict__ kpad,  // [B,H,S2W,DH]
                 const bf16_t* __restrict__ vpT,   // [B,H,DH,S2W]
                 const float* __restrict__ keybias,
                 bf16_t* __restrict__ ao_b) {      // [B,S,H*DH]
  __shared__ float sc[32][392];
  __shared__ float pgs[32];
  int blk = blockIdx.x;
  int qq = blk & 3; blk >>= 2;
  int n  = blk % NB_; blk /= NB_;
  int h  = blk % H_;
  int b  = blk / H_;
  int wave = threadIdx.x >> 5, lane = threadIdx.x & 31;
  int lo = lane & 15, hi = lane >> 4;
  int s0 = n * W_ + qq * 32;
  long qbase = ((long)(b * S_ + s0)) * D_ + h * DH_;
  long kbase = (((long)(b * H_ + h)) * S2W_ + (long)n * W_) * DH_;

  // phase 1: scores 32x384 = 48 16x16 tiles, 6 per wave, K=64 (2 WMMA each)
  for (int t = wave; t < 48; t += 8) {
    int mi = t / 24, nti = t % 24;
    v8f acc = {};
    {
      v16bf a  = load_frag(q_b + qbase + (long)mi * 16 * D_ + 0, D_);
      v16bf bb = load_frag(kpad + kbase + (long)nti * 16 * DH_ + 0, DH_);
      acc = wmma_bf16(a, bb, acc);
    }
    {
      v16bf a  = load_frag(q_b + qbase + (long)mi * 16 * D_ + 32, D_);
      v16bf bb = load_frag(kpad + kbase + (long)nti * 16 * DH_ + 32, DH_);
      acc = wmma_bf16(a, bb, acc);
    }
    int jj = nti * 16 + lo;
#pragma unroll
    for (int r = 0; r < 8; ++r) {
      int qi = mi * 16 + r + 8 * hi;          // 0..31 local row
      int ib = qq * 32 + qi;                  // query index within W block
      int pos = n * W_ - W_ + jj;             // absolute key position
      float bias;
      if (jj < ib || jj > ib + 2 * W_ || pos < 0 || pos >= S_) bias = -1e9f;
      else bias = keybias[(long)b * S_ + pos];
      sc[qi][jj] = acc[r] + bias;
    }
  }
  __syncthreads();

  // phase 2: per-row softmax over [global col, 384 band cols]; 4 rows per wave
  for (int rr = 0; rr < 4; ++rr) {
    int qi = wave * 4 + rr;
    const bf16_t* qr = q_b + qbase + (long)qi * D_;
    const bf16_t* kg = kpad + (((long)(b * H_ + h)) * S2W_ + W_) * DH_;  // key pos 0
    float part = (float)qr[lane * 2]     * (float)kg[lane * 2] +
                 (float)qr[lane * 2 + 1] * (float)kg[lane * 2 + 1];
#pragma unroll
    for (int m = 16; m > 0; m >>= 1) part += __shfl_xor(part, m, 32);
    float sg = part;
    float mx = sg;
    for (int j = lane; j < 384; j += 32) mx = fmaxf(mx, sc[qi][j]);
#pragma unroll
    for (int m = 16; m > 0; m >>= 1) mx = fmaxf(mx, __shfl_xor(mx, m, 32));
    float sum = 0.f;
    for (int j = lane; j < 384; j += 32) sum += __expf(sc[qi][j] - mx);
#pragma unroll
    for (int m = 16; m > 0; m >>= 1) sum += __shfl_xor(sum, m, 32);
    sum += __expf(sg - mx);
    float inv = 1.f / sum;
    if (lane == 0) pgs[qi] = __expf(sg - mx) * inv;
    // overwrite scores with bf16 probabilities in place (writes trail reads)
    bf16_t* pp = (bf16_t*)&sc[qi][0];
    for (int j = lane; j < 384; j += 32) {
      float p = __expf(sc[qi][j] - mx) * inv;
      pp[j] = (bf16_t)p;
    }
  }
  __syncthreads();

  // phase 3: out(32x64) = P(32x384) @ Vband(384x64) + pg*vg; 8 tiles, 1/wave
  {
    int mi = wave >> 2, di = wave & 3;
    v8f acc = {};
    const bf16_t* pbase = (const bf16_t*)&sc[0][0] + (long)mi * 16 * 784;  // ld=784
    long vbase = ((long)(b * H_ + h) * DH_ + di * 16) * S2W_ + (long)n * W_;
#pragma unroll
    for (int k = 0; k < 384; k += 32) {
      v16bf a  = load_frag(pbase + k, 784);
      v16bf bb = load_frag(vpT + vbase + k, S2W_);
      acc = wmma_bf16(a, bb, acc);
    }
    int d = di * 16 + lo;
    float vg = (float)vpT[((long)(b * H_ + h) * DH_ + d) * S2W_ + W_];  // v pos 0
#pragma unroll
    for (int r = 0; r < 8; ++r) {
      int qi = mi * 16 + r + 8 * hi;
      float o = acc[r] + pgs[qi] * vg;
      ao_b[((long)(b * S_ + s0 + qi)) * D_ + h * DH_ + d] = (bf16_t)o;
    }
  }
}

// ---------------- full attention for the global token (row s==0) ----------------
__global__ void __launch_bounds__(256)
gattn_kernel(const bf16_t* __restrict__ q_b, const bf16_t* __restrict__ k_b,
             const bf16_t* __restrict__ v_b, const float* __restrict__ gbias,
             bf16_t* __restrict__ ao_b) {
  __shared__ float sc4[S_];
  __shared__ float red[256];
  __shared__ float qsh[DH_];
  int bh = blockIdx.x;
  int b = bh / H_, h = bh % H_;
  int tid = threadIdx.x;
  if (tid < DH_) qsh[tid] = (float)q_b[((long)b * S_) * D_ + h * DH_ + tid];
  __syncthreads();
  float lmax = -1e30f;
  for (int j = tid; j < S_; j += 256) {
    const bf16_t* kr = k_b + ((long)(b * S_ + j)) * D_ + h * DH_;
    float s = 0.f;
    for (int d0 = 0; d0 < DH_; ++d0) s += qsh[d0] * (float)kr[d0];
    s += gbias[(long)b * S_ + j];
    sc4[j] = s;
    lmax = fmaxf(lmax, s);
  }
  red[tid] = lmax; __syncthreads();
  for (int st = 128; st > 0; st >>= 1) {
    if (tid < st) red[tid] = fmaxf(red[tid], red[tid + st]);
    __syncthreads();
  }
  float mx = red[0]; __syncthreads();
  float lsum = 0.f;
  for (int j = tid; j < S_; j += 256) {
    float e = __expf(sc4[j] - mx);
    sc4[j] = e;
    lsum += e;
  }
  red[tid] = lsum; __syncthreads();
  for (int st = 128; st > 0; st >>= 1) {
    if (tid < st) red[tid] += red[tid + st];
    __syncthreads();
  }
  float inv = 1.f / red[0]; __syncthreads();
  int d = tid & 63, grp = tid >> 6;
  float acc = 0.f;
  for (int j = grp; j < S_; j += 4)
    acc += sc4[j] * (float)v_b[((long)(b * S_ + j)) * D_ + h * DH_ + d];
  red[tid] = acc; __syncthreads();
  if (grp == 0) {
    float o = (red[d] + red[64 + d] + red[128 + d] + red[192 + d]) * inv;
    ao_b[((long)b * S_) * D_ + h * DH_ + d] = (bf16_t)o;
  }
}

// ---------------- residual + LayerNorm (row of 768 = 256 threads x 3) ----------------
__global__ void __launch_bounds__(256)
ln_kernel(const float* __restrict__ resid, const float* __restrict__ delta,
          const float* __restrict__ g, const float* __restrict__ beta,
          float* __restrict__ out_f, bf16_t* __restrict__ out_b) {
  int row = blockIdx.x;
  const float* r0 = resid + (long)row * D_;
  const float* d0 = delta ? delta + (long)row * D_ : nullptr;
  float vals[3];
  float s = 0.f, s2 = 0.f;
#pragma unroll
  for (int i = 0; i < 3; ++i) {
    int c = threadIdx.x + i * 256;
    float v = r0[c] + (d0 ? d0[c] : 0.f);
    vals[i] = v; s += v; s2 += v * v;
  }
  __shared__ float sh[2][8];
#pragma unroll
  for (int m = 16; m > 0; m >>= 1) { s += __shfl_xor(s, m, 32); s2 += __shfl_xor(s2, m, 32); }
  int wv = threadIdx.x >> 5, ln = threadIdx.x & 31;
  if (ln == 0) { sh[0][wv] = s; sh[1][wv] = s2; }
  __syncthreads();
  if (wv == 0) {
    float a = ln < 8 ? sh[0][ln] : 0.f;
    float b2 = ln < 8 ? sh[1][ln] : 0.f;
#pragma unroll
    for (int m = 4; m > 0; m >>= 1) { a += __shfl_xor(a, m, 32); b2 += __shfl_xor(b2, m, 32); }
    if (ln == 0) { sh[0][0] = a; sh[1][0] = b2; }
  }
  __syncthreads();
  float mean = sh[0][0] / D_;
  float var  = sh[1][0] / D_ - mean * mean;
  float inv  = rsqrtf(var + 1e-5f);
#pragma unroll
  for (int i = 0; i < 3; ++i) {
    int c = threadIdx.x + i * 256;
    float y = (vals[i] - mean) * inv * g[c] + beta[c];
    if (out_f) out_f[(long)row * D_ + c] = y;
    if (out_b) out_b[(long)row * D_ + c] = (bf16_t)y;
  }
}

// ---------------- host orchestration ----------------
extern "C" void kernel_launch(void* const* d_in, const int* in_sizes, int n_in,
                              void* d_out, int out_size, void* d_ws, size_t ws_size,
                              hipStream_t stream) {
  (void)in_sizes; (void)n_in; (void)out_size; (void)ws_size;
  const int*   call     = (const int*)d_in[0];
  const unsigned char* pad_mask = (const unsigned char*)d_in[7];
  const float* emb   = (const float*)d_in[9];
  const float* Wq    = (const float*)d_in[10];
  const float* bq    = (const float*)d_in[11];
  const float* Wk    = (const float*)d_in[12];
  const float* bk    = (const float*)d_in[13];
  const float* Wv    = (const float*)d_in[14];
  const float* bv    = (const float*)d_in[15];
  const float* Wo    = (const float*)d_in[16];
  const float* bo    = (const float*)d_in[17];
  const float* W1    = (const float*)d_in[18];
  const float* b1    = (const float*)d_in[19];
  const float* W2    = (const float*)d_in[20];
  const float* b2    = (const float*)d_in[21];
  const float* g1    = (const float*)d_in[22];
  const float* beta1 = (const float*)d_in[23];
  const float* g2    = (const float*)d_in[24];
  const float* beta2 = (const float*)d_in[25];
  const float* gf    = (const float*)d_in[26];
  const float* betaf = (const float*)d_in[27];
  const float* Wc    = (const float*)d_in[28];
  const float* bc    = (const float*)d_in[29];
  float* out = (float*)d_out;

  char* ws = (char*)d_ws;
  size_t off = 0;
  auto alloc = [&](size_t bytes) -> void* {
    void* p = ws + off;
    off += (bytes + 255) & ~(size_t)255;
    return p;
  };
  bf16_t* xb   = (bf16_t*)alloc(MS_ * D_ * sizeof(bf16_t));
  float*  xf   = (float*) alloc(MS_ * D_ * sizeof(float));
  bf16_t* q_b  = (bf16_t*)alloc(MS_ * D_ * sizeof(bf16_t));
  bf16_t* k_b  = (bf16_t*)alloc(MS_ * D_ * sizeof(bf16_t));
  bf16_t* v_b  = (bf16_t*)alloc(MS_ * D_ * sizeof(bf16_t));
  bf16_t* kpad = (bf16_t*)alloc((long)B_ * H_ * S2W_ * DH_ * sizeof(bf16_t));
  bf16_t* vpT  = (bf16_t*)alloc((long)B_ * H_ * S2W_ * DH_ * sizeof(bf16_t));
  bf16_t* ao_b = (bf16_t*)alloc(MS_ * D_ * sizeof(bf16_t));
  float*  tmpf = (float*) alloc(MS_ * D_ * sizeof(float));
  bf16_t* h_b  = (bf16_t*)alloc(MS_ * FF_ * sizeof(bf16_t));
  bf16_t* wqb  = (bf16_t*)alloc((long)D_ * D_ * sizeof(bf16_t));
  bf16_t* wkb  = (bf16_t*)alloc((long)D_ * D_ * sizeof(bf16_t));
  bf16_t* wvb  = (bf16_t*)alloc((long)D_ * D_ * sizeof(bf16_t));
  bf16_t* wob  = (bf16_t*)alloc((long)D_ * D_ * sizeof(bf16_t));
  bf16_t* w1b  = (bf16_t*)alloc((long)FF_ * D_ * sizeof(bf16_t));
  bf16_t* w2b  = (bf16_t*)alloc((long)D_ * FF_ * sizeof(bf16_t));
  bf16_t* wcb  = (bf16_t*)alloc((long)V_ * D_ * sizeof(bf16_t));
  float*  keybias = (float*)alloc(MS_ * sizeof(float));
  float*  gbias   = (float*)alloc(MS_ * sizeof(float));

  const float rscale = 0.125f;  // 1/sqrt(64)

  // embedding + masks
  embed_kernel<<<(int)((MS_ * D_) / 256), 256, 0, stream>>>(call, emb, xf, xb);
  bias_kernel<<<(int)(MS_ / 256), 256, 0, stream>>>(pad_mask, keybias, gbias);

  auto cvt = [&](const float* src, bf16_t* dst, long n) {
    cvt_bf16_kernel<<<(int)((n + 255) / 256), 256, 0, stream>>>(src, dst, n);
  };
  auto gemm = [&](const bf16_t* A, const bf16_t* Wt, const float* bias, float alpha,
                  int relu, float* Cf, bf16_t* Cb, int M, int N, int K) {
    dim3 grid(M / 128, N / 64);
    gemm_bf16_kernel<<<grid, 256, 0, stream>>>(A, Wt, bias, alpha, relu, Cf, Cb, M, N, K);
  };

  for (int l = 0; l < L_; ++l) {
    const long dd = (long)D_ * D_;
    cvt(Wq + l * dd, wqb, dd);
    cvt(Wk + l * dd, wkb, dd);
    cvt(Wv + l * dd, wvb, dd);
    cvt(Wo + l * dd, wob, dd);
    cvt(W1 + (long)l * FF_ * D_, w1b, (long)FF_ * D_);
    cvt(W2 + (long)l * D_ * FF_, w2b, (long)D_ * FF_);

    // Q (pre-scaled), K, V
    gemm(xb, wqb, bq + l * D_, rscale, 0, nullptr, q_b, (int)MS_, D_, D_);
    gemm(xb, wkb, bk + l * D_, 1.f,    0, nullptr, k_b, (int)MS_, D_, D_);
    gemm(xb, wvb, bv + l * D_, 1.f,    0, nullptr, v_b, (int)MS_, D_, D_);

    // pad K, transpose+pad V
    long npad = (long)B_ * H_ * S2W_ * DH_;
    padkv_kernel<<<(int)(npad / 256), 256, 0, stream>>>(k_b, v_b, kpad, vpT);

    // banded attention + global column
    band_attn_kernel<<<B_ * H_ * NB_ * 4, 256, 0, stream>>>(q_b, kpad, vpT, keybias, ao_b);
    // global token full attention (overwrites row s==0)
    gattn_kernel<<<B_ * H_, 256, 0, stream>>>(q_b, k_b, v_b, gbias, ao_b);

    // output projection -> tmpf, then LN1 (residual)
    gemm(ao_b, wob, bo + l * D_, 1.f, 0, tmpf, nullptr, (int)MS_, D_, D_);
    ln_kernel<<<(int)MS_, 256, 0, stream>>>(xf, tmpf, g1 + l * D_, beta1 + l * D_, xf, xb);

    // FFN
    gemm(xb, w1b, b1 + l * FF_, 1.f, 1, nullptr, h_b, (int)MS_, FF_, D_);
    gemm(h_b, w2b, b2 + l * D_, 1.f, 0, tmpf, nullptr, (int)MS_, D_, FF_);
    ln_kernel<<<(int)MS_, 256, 0, stream>>>(xf, tmpf, g2 + l * D_, beta2 + l * D_, xf, xb);
  }

  // final LN -> bf16, classifier -> d_out (f32)
  ln_kernel<<<(int)MS_, 256, 0, stream>>>(xf, nullptr, gf, betaf, nullptr, xb);
  cvt(Wc, wcb, (long)V_ * D_);
  gemm(xb, wcb, bc, 1.f, 0, out, nullptr, (int)MS_, V_, D_);
}
